// LocalAttention_39281770889542
// MI455X (gfx1250) — matrix operand order repaired and checked by reading, not compiled, not run on hardware
//
#include <hip/hip_runtime.h>
#include <hip/hip_bf16.h>

typedef __attribute__((ext_vector_type(16))) _Float16 v16h;
typedef __attribute__((ext_vector_type(8)))  _Float16 v8h;
typedef __attribute__((ext_vector_type(4)))  _Float16 v4h;
typedef __attribute__((ext_vector_type(8)))  float    v8f;

#define T_SEQ    4096
#define EDIM     64
#define BUCKET   256
#define WIN      512
#define NBUCKETS (T_SEQ / BUCKET)   // 16

// 1/sqrt(64) * log2(e): scores computed directly in log2 domain so softmax
// uses bare v_exp_f32 (natively 2^x) with no multiply.
#define QSCALE 0.1803368801111204f

__device__ __forceinline__ float fast_exp2(float x) {
  return __builtin_amdgcn_exp2f(x);
}

__device__ __forceinline__ v8f wmma_f16(v16h a, v16h b, v8f c) {
  // D = A(16x32 f16) * B(32x16 f16) + C(16x16 f32)
  return __builtin_amdgcn_wmma_f32_16x16x32_f16(false, a, false, b, (short)0, c,
                                                false, false);
}

// A-operand fragment (16x32 f16): lanes 0-15 hold K0-7,K16-23;
// lanes 16-31 hold K8-15,K24-31 (hlf = lane>>4).
__device__ __forceinline__ v16h load_fragA(const _Float16* rowK, int hlf) {
  const v8h* p0 = (const v8h*)(rowK + hlf * 8);
  const v8h* p1 = (const v8h*)(rowK + hlf * 8 + 16);
  v8h x = *p0, y = *p1;
  v16h r;
#pragma unroll
  for (int i = 0; i < 8; ++i) { r[i] = x[i]; r[i + 8] = y[i]; }
  return r;
}

// B-operand fragment (32x16 f16): lanes 0-15 hold K0-7,K16-23;
// lanes 16-31 hold K8-15,K24-31.
__device__ __forceinline__ v16h load_fragB(const _Float16* colK, int hlf) {
  const v8h* p0 = (const v8h*)(colK + hlf * 8);
  const v8h* p1 = (const v8h*)(colK + hlf * 8 + 16);
  v8h x = *p0, y = *p1;
  v16h r;
#pragma unroll
  for (int i = 0; i < 8; ++i) { r[i] = x[i]; r[i + 8] = y[i]; }
  return r;
}

// new_m = max(m_old, max over 16 lanes of max(s0..s3)), entirely in asm:
// tree max (max3+max), 4x DPP16 ROW_XMASK hops, merge with m_old.
__device__ __forceinline__ float rowmax_update(float s0, float s1, float s2,
                                               float s3, float mold) {
  float t, u;
  asm("v_max3_num_f32 %0, %1, %2, %3" : "=v"(t) : "v"(s0), "v"(s1), "v"(s2));
  asm("v_max_num_f32_e32 %0, %1, %2" : "=v"(u) : "v"(t), "v"(s3));
  asm("v_max_num_f32_dpp %0, %1, %1 row_xmask:1 row_mask:0xf bank_mask:0xf"
      : "=v"(t) : "v"(u));
  asm("v_max_num_f32_dpp %0, %1, %1 row_xmask:2 row_mask:0xf bank_mask:0xf"
      : "=v"(u) : "v"(t));
  asm("v_max_num_f32_dpp %0, %1, %1 row_xmask:4 row_mask:0xf bank_mask:0xf"
      : "=v"(t) : "v"(u));
  asm("v_max_num_f32_dpp %0, %1, %1 row_xmask:8 row_mask:0xf bank_mask:0xf"
      : "=v"(u) : "v"(t));
  asm("v_max_num_f32_e32 %0, %1, %2" : "=v"(t) : "v"(u), "v"(mold));
  return t;
}

// Broadcast lane 0 of each 16-lane row to the whole row (DPP ROW_SHARE).
__device__ __forceinline__ float row_bcast0(float v) {
  float r;
  asm("v_mov_b32_dpp %0, %1 row_share:0 row_mask:0xf bank_mask:0xf"
      : "=v"(r) : "v"(v));
  return r;
}

// One 64-key chunk: 4 score subtiles, one online-softmax update, O/L accum.
// Key contraction axis uses permuted order p = 4*(k%16) + k/16 inside the
// chunk (consistently in P scratch and Vt), making each lane's P stores a
// single b64. The permutation is invariant for the P*V contraction.
template <bool MASKED>
__device__ __forceinline__ void process64(
    int kk, int wave, int hlf, int col, const _Float16* Ks, const _Float16* Vt,
    _Float16* myP, const v16h& qa0, const v16h& qa1, const v16h& onesB,
    float* mrow, v8f* o) {
  // ---- S = Q * K^T : four 16-key subtiles, K-dim in 2 chunks of 32 ----
  v8f s[4];
#pragma unroll
  for (int t = 0; t < 4; ++t) {
    const _Float16* kc = Ks + (size_t)(kk + t * 16 + col) * EDIM;
    v8f acc = {};
    acc = wmma_f16(qa0, load_fragB(kc,      hlf), acc);
    acc = wmma_f16(qa1, load_fragB(kc + 32, hlf), acc);
    s[t] = acc;
  }

  // ---- (mask +) online softmax in log2 domain ----
  float scl[8];
#pragma unroll
  for (int i = 0; i < 8; ++i) {
    if (MASKED) {
      const int qw = BUCKET + wave * 16 + hlf * 8 + i;  // query window pos
#pragma unroll
      for (int t = 0; t < 4; ++t)
        s[t][i] = (qw < kk + t * 16 + col) ? -1e30f : s[t][i];
    }
    const float mnew = rowmax_update(s[0][i], s[1][i], s[2][i], s[3][i], mrow[i]);
    scl[i] = fast_exp2(mrow[i] - mnew);
    mrow[i] = mnew;
#pragma unroll
    for (int t = 0; t < 4; ++t) s[t][i] = fast_exp2(s[t][i] - mnew);
  }
#pragma unroll
  for (int i = 0; i < 8; ++i) {
    o[0][i] *= scl[i]; o[1][i] *= scl[i]; o[2][i] *= scl[i];
    o[3][i] *= scl[i]; o[4][i] *= scl[i];
  }

  // ---- P (C-layout -> A-layout): permuted columns make stores contiguous ----
#pragma unroll
  for (int i = 0; i < 8; ++i) {
    const int r = hlf * 8 + i;
    v4h pk;
    pk[0] = (_Float16)s[0][i]; pk[1] = (_Float16)s[1][i];
    pk[2] = (_Float16)s[2][i]; pk[3] = (_Float16)s[3][i];
    *(v4h*)&myP[r * 64 + 4 * col] = pk;   // one b64 per row
  }
  asm volatile("s_wait_dscnt 0" ::: "memory");  // per-wave LDS RAW fence
  v16h pA0 = load_fragA(myP + col * 64,      hlf);
  v16h pA1 = load_fragA(myP + col * 64 + 32, hlf);

  // ---- O += P*V ; L += P*ones (running softmax denominator in col 0) ----
#pragma unroll
  for (int n = 0; n < 4; ++n) {
    const _Float16* vc = Vt + (size_t)(n * 16 + col) * WIN + kk;
    o[n] = wmma_f16(pA0, load_fragB(vc,      hlf), o[n]);
    o[n] = wmma_f16(pA1, load_fragB(vc + 32, hlf), o[n]);
  }
  o[4] = wmma_f16(pA0, onesB, o[4]);
  o[4] = wmma_f16(pA1, onesB, o[4]);
}

__global__ __launch_bounds__(512)
void local_attn_fa_kernel(const float* __restrict__ Q,
                          const float* __restrict__ K,
                          const float* __restrict__ V,
                          float* __restrict__ O) {
  extern __shared__ _Float16 smem[];
  _Float16* Ks = smem;                 // [WIN][EDIM] row-major f16
  _Float16* Vt = Ks + WIN * EDIM;      // [EDIM][WIN] transposed f16 (permuted keys)
  _Float16* Ps = Vt + EDIM * WIN;      // [16 waves][16][64] P scratch

  const int blk    = blockIdx.x;
  const int bh     = blk >> 4;               // 0..31 (batch*heads)
  const int bucket = blk & (NBUCKETS - 1);   // 0..15
  const int q0     = bucket * BUCKET;
  const size_t base = (size_t)bh * T_SEQ * EDIM;

  // ---------- Phase 1: stage K (row-major) and V (transposed) as f16 ----------
  {
    const int t   = threadIdx.x;           // 0..511
    // K: one 64-dim key row per thread, vectorized b128 stores.
    const int kwl = q0 - BUCKET + t;       // local key index; <0 => pad
    v8h* krow = (v8h*)(Ks + (size_t)t * EDIM);
    if (kwl >= 0) {
      const float4* ksrc = (const float4*)(K + base + (size_t)kwl * EDIM);
#pragma unroll
      for (int i = 0; i < 8; ++i) {
        float4 f0 = ksrc[2 * i], f1 = ksrc[2 * i + 1];
        v8h h;
        h[0] = (_Float16)f0.x; h[1] = (_Float16)f0.y;
        h[2] = (_Float16)f0.z; h[3] = (_Float16)f0.w;
        h[4] = (_Float16)f1.x; h[5] = (_Float16)f1.y;
        h[6] = (_Float16)f1.z; h[7] = (_Float16)f1.w;
        krow[i] = h;
      }
    } else {
      v8h z = {};
#pragma unroll
      for (int i = 0; i < 8; ++i) krow[i] = z;
    }
    // V: 4-key x 16-dim patch per thread; original keys 64*cc + c + {0,16,32,48}
    // are contiguous in permuted space -> packed b64 transposed stores.
    const int G    = t & 127;              // permuted 4-group 0..127
    const int cc   = G >> 4;               // 64-key chunk 0..7
    const int c    = G & 15;               // column-in-chunk 0..15
    const int dh   = (t >> 7) * 16;        // dim base 0,16,32,48
    const int kwl0 = q0 - BUCKET + cc * 64 + c;  // pad status uniform per chunk
    float vals[4][16];
    if (kwl0 >= 0) {
#pragma unroll
      for (int ky = 0; ky < 4; ++ky) {
        const float4* vsrc =
            (const float4*)(V + base + (size_t)(kwl0 + 16 * ky) * EDIM + dh);
#pragma unroll
        for (int j = 0; j < 4; ++j) {
          float4 f = vsrc[j];
          vals[ky][4 * j]     = f.x; vals[ky][4 * j + 1] = f.y;
          vals[ky][4 * j + 2] = f.z; vals[ky][4 * j + 3] = f.w;
        }
      }
    } else {
#pragma unroll
      for (int ky = 0; ky < 4; ++ky)
#pragma unroll
        for (int j = 0; j < 16; ++j) vals[ky][j] = 0.f;
    }
#pragma unroll
    for (int d = 0; d < 16; ++d) {
      v4h pk;
      pk[0] = (_Float16)vals[0][d]; pk[1] = (_Float16)vals[1][d];
      pk[2] = (_Float16)vals[2][d]; pk[3] = (_Float16)vals[3][d];
      *(v4h*)&Vt[(size_t)(dh + d) * WIN + cc * 64 + 4 * c] = pk;
    }
  }
  __syncthreads();

  // ---------- Phase 2: flash attention, one 16-query tile per wave ----------
  const int lane = threadIdx.x & 31;
  const int wave = threadIdx.x >> 5;   // M-tile 0..15
  const int hlf  = lane >> 4;
  const int col  = lane & 15;

  // Q fragments, QSCALE (1/sqrt(64)*log2e) folded in. A-layout row = col.
  v16h qa0, qa1;
  {
    const float* qr = Q + base + (size_t)(q0 + wave * 16 + col) * EDIM;
    v16h a;
#pragma unroll
    for (int i = 0; i < 8; ++i) a[i]     = (_Float16)(qr[hlf * 8 + i]      * QSCALE);
#pragma unroll
    for (int i = 0; i < 8; ++i) a[i + 8] = (_Float16)(qr[hlf * 8 + 16 + i] * QSCALE);
    qa0 = a;
#pragma unroll
    for (int i = 0; i < 8; ++i) a[i]     = (_Float16)(qr[32 + hlf * 8 + i]      * QSCALE);
#pragma unroll
    for (int i = 0; i < 8; ++i) a[i + 8] = (_Float16)(qr[32 + hlf * 8 + 16 + i] * QSCALE);
    qa1 = a;
  }

  // B_ones: column 0 all ones, other columns zero -> row-sum accumulator.
  v16h onesB;
  {
    const _Float16 ov = (col == 0) ? (_Float16)1.f : (_Float16)0.f;
#pragma unroll
    for (int i = 0; i < 16; ++i) onesB[i] = ov;
  }

  float mrow[8];
  v8f o[5] = {};
#pragma unroll
  for (int i = 0; i < 8; ++i) mrow[i] = -1e30f;

  _Float16* myP = Ps + wave * (16 * 64);
  const int qw_min = BUCKET + wave * 16;          // first query's window pos
  const int qwmax  = qw_min + 15;                 // last query's window pos
  const int kstart = (bucket == 0) ? BUCKET : 0;  // skip fully-padded chunks

  int kk = kstart;
  // Interior chunks: every key pos <= qw_min -> never masked.
  for (; kk + 63 <= qw_min; kk += 64)
    process64<false>(kk, wave, hlf, col, Ks, Vt, myP, qa0, qa1, onesB, mrow, o);
  // Diagonal chunk: elementwise causal mask (fully-masked subtiles give p=0).
  for (; kk <= qwmax; kk += 64)
    process64<true>(kk, wave, hlf, col, Ks, Vt, myP, qa0, qa1, onesB, mrow, o);

  // ---------- epilogue: broadcast denominator, normalize, store ----------
#pragma unroll
  for (int i = 0; i < 8; ++i) {
    const float l   = row_bcast0(o[4][i]);  // col-0 holds the row sum
    const float inv = 1.0f / l;
    const int r = hlf * 8 + i;
    float* orow = O + base + (size_t)(q0 + wave * 16 + r) * EDIM;
    orow[col]      = o[0][i] * inv;
    orow[16 + col] = o[1][i] * inv;
    orow[32 + col] = o[2][i] * inv;
    orow[48 + col] = o[3][i] * inv;
  }
}

extern "C" void kernel_launch(void* const* d_in, const int* in_sizes, int n_in,
                              void* d_out, int out_size, void* d_ws, size_t ws_size,
                              hipStream_t stream) {
  const float* q = (const float*)d_in[0];
  const float* k = (const float*)d_in[1];
  const float* v = (const float*)d_in[2];
  float* out = (float*)d_out;
  // 2*16 (bh) * 16 buckets = 512 workgroups; 512 threads = 16 wave32 per WG.
  // LDS: 64KB K + 64KB Vt + 32KB P scratch = 160KB -> 2 WGs per 320KB WGP.
  const size_t shmem =
      (size_t)(WIN * EDIM + EDIM * WIN + 16 * 16 * 64) * sizeof(_Float16);
  local_attn_fa_kernel<<<dim3(32 * NBUCKETS), dim3(512), shmem, stream>>>(q, k, v, out);
}
